// TGN_58815282152018
// MI455X (gfx1250) — matrix operand order repaired and checked by reading, not compiled, not run on hardware
//
#include <hip/hip_runtime.h>
#include <hip/hip_bf16.h>
#include <math.h>

typedef __attribute__((ext_vector_type(16))) _Float16 v16h;
typedef __attribute__((ext_vector_type(8)))  _Float16 v8h;
typedef __attribute__((ext_vector_type(4)))  _Float16 v4h;
typedef __attribute__((ext_vector_type(8)))  float    v8f;

#define B_   256
#define N_   20
#define N2_  400
#define D_   128
#define ND_  172
#define GF_  1261
#define CB   32
#define NCHUNK (B_/CB)
#define ROWS_N (CB*(1+N_+N2_))   // 13472
#define ROWS_E (CB*(N_+N2_))     // 13440
#define ROWS2  (CB*N2_)          // 12800
#define ROWS1  (CB*N_)           // 640
#define INV_SQRT_DH 0.08838834764831845f  // 1/sqrt(128)

// ---------------------------------------------------------------------------
// WMMA GEMM: act(X[M,Kpad](f16) @ Wfrag + bias) -> Yh (f16) and/or Yf (f32)
// one wave computes a 16x64 tile: 1 A-fragment reused over 4 B-fragments.
// ---------------------------------------------------------------------------
__global__ __launch_bounds__(32)
void k_gemm(const _Float16* __restrict__ X, const _Float16* __restrict__ Bfrag,
            const float* __restrict__ bias, _Float16* __restrict__ Yh,
            float* __restrict__ Yf, int M, int Kpad, int Ncol, int relu) {
  const int m0   = blockIdx.x * 16;
  const int nt0  = blockIdx.y * 4;
  const int lane = threadIdx.x;
  const int half = lane >> 4;
  const int l15  = lane & 15;

  int rowA = m0 + l15;
  if (rowA >= M) rowA = M - 1;
  const _Float16* xr = X + (size_t)rowA * Kpad;

  const int nkt    = Kpad >> 5;
  const int ntiles = Ncol >> 4;
  v8f acc[4] = {{}, {}, {}, {}};

  for (int kc = 0; kc < nkt; ++kc) {
    const int kb = kc * 32 + half * 8;
    v8h alo = *(const v8h*)(xr + kb);
    v8h ahi = *(const v8h*)(xr + kb + 16);
    v16h a = __builtin_shufflevector(alo, ahi, 0, 1, 2, 3, 4, 5, 6, 7,
                                     8, 9, 10, 11, 12, 13, 14, 15);
    const _Float16* bp = Bfrag + (((size_t)kc * ntiles + nt0) * 32 + lane) * 16;
    v16h b0 = *(const v16h*)(bp);
    v16h b1 = *(const v16h*)(bp + 512);
    v16h b2 = *(const v16h*)(bp + 1024);
    v16h b3 = *(const v16h*)(bp + 1536);
    acc[0] = __builtin_amdgcn_wmma_f32_16x16x32_f16(false, a, false, b0, (short)0, acc[0], false, false);
    acc[1] = __builtin_amdgcn_wmma_f32_16x16x32_f16(false, a, false, b1, (short)0, acc[1], false, false);
    acc[2] = __builtin_amdgcn_wmma_f32_16x16x32_f16(false, a, false, b2, (short)0, acc[2], false, false);
    acc[3] = __builtin_amdgcn_wmma_f32_16x16x32_f16(false, a, false, b3, (short)0, acc[3], false, false);
  }

#pragma unroll
  for (int c = 0; c < 4; ++c) {
    const int col = (nt0 + c) * 16 + l15;
    const float bv = bias ? bias[col] : 0.0f;
#pragma unroll
    for (int r = 0; r < 8; ++r) {
      const int row = m0 + r + half * 8;
      if (row < M) {
        float v = acc[c][r] + bv;
        if (relu) v = fmaxf(v, 0.0f);
        if (Yh) Yh[(size_t)row * Ncol + col] = (_Float16)v;
        if (Yf) Yf[(size_t)row * Ncol + col] = v;
      }
    }
  }
}

// Pack fp32 row-major W[K,Ncol] into WMMA B-fragment order (zero pad K->Kpad).
__global__ void k_pack_b(const float* __restrict__ W, _Float16* __restrict__ out,
                         int K, int Kpad, int Ncol) {
  const int total = (Kpad / 32) * (Ncol / 16) * 32;
  int t = blockIdx.x * blockDim.x + threadIdx.x;
  if (t >= total) return;
  const int lane = t & 31;
  const int ft   = t >> 5;
  const int ntiles = Ncol / 16;
  const int nt = ft % ntiles;
  const int kc = ft / ntiles;
  const int half = lane >> 4;
  const int n = nt * 16 + (lane & 15);
  _Float16* o = out + (size_t)t * 16;
#pragma unroll
  for (int j = 0; j < 16; ++j) {
    const int k = kc * 32 + half * 8 + (j < 8 ? j : j + 8);
    o[j] = (_Float16)((k < K) ? W[(size_t)k * Ncol + n] : 0.0f);
  }
}

// ---------------------------------------------------------------------------
// Gather / build kernels (all activation outputs f16)
// ---------------------------------------------------------------------------
__device__ __forceinline__ v4h cvt4(float a, float b, float c, float d) {
  v4h v; v[0] = (_Float16)a; v[1] = (_Float16)b; v[2] = (_Float16)c; v[3] = (_Float16)d;
  return v;
}

__global__ void k_smax(const float* __restrict__ gf, float* __restrict__ smaxp,
                       float* __restrict__ outp) {
  int b = blockIdx.x * blockDim.x + threadIdx.x;
  if (b >= B_) return;
  const float* p = gf + (size_t)b * GF_ + 841;
  float m = p[0];
  for (int c = 1; c < N_ + N2_; ++c) m = fmaxf(m, p[c]);
  smaxp[b] = m;
  outp[b]  = m;
}

__global__ void k_gather_node(const float* __restrict__ tbl, const int* __restrict__ srcid,
                              const float* __restrict__ gf, _Float16* __restrict__ out, int b0) {
  int t = blockIdx.x * blockDim.x + threadIdx.x;
  if (t >= ROWS_N * 48) return;
  const int r = t / 48, c4 = t % 48;
  int id;
  if (r < CB) {
    id = srcid[b0 + r];
  } else if (r < CB * 21) {
    const int rr = r - CB, b = rr / N_, i = rr % N_;
    id = (int)gf[(size_t)(b0 + b) * GF_ + 1 + i];
  } else {
    const int rr = r - CB * 21, b = rr / N2_, j = rr % N2_;
    id = (int)gf[(size_t)(b0 + b) * GF_ + 21 + j];
  }
  v4h h = {};
  if (c4 < 43) {
    float4 v = *(const float4*)(tbl + (size_t)id * ND_ + c4 * 4);
    h = cvt4(v.x, v.y, v.z, v.w);
  }
  *(v4h*)(out + (size_t)r * 192 + c4 * 4) = h;
}

__global__ void k_gather_edge(const float* __restrict__ tbl, const float* __restrict__ e1o,
                              const float* __restrict__ e2o, _Float16* __restrict__ out, int b0) {
  int t = blockIdx.x * blockDim.x + threadIdx.x;
  if (t >= ROWS_E * 8) return;
  const int r = t / 8, c4 = t % 8;
  int id;
  if (r < ROWS1) {
    const int b = r / N_, i = r % N_;
    id = (int)e1o[((size_t)(b0 + b) * N_ + i) * 9];
  } else {
    const int rr = r - ROWS1, b = rr / N2_, j = rr % N2_;
    id = (int)e2o[((size_t)(b0 + b) * N2_ + j) * 9];
  }
  v4h h = {};
  if (c4 < 4) {
    float4 v = *(const float4*)(tbl + (size_t)id * 16 + c4 * 4);
    h = cvt4(v.x, v.y, v.z, v.w);
  }
  *(v4h*)(out + (size_t)r * 32 + c4 * 4) = h;
}

__global__ void k_build_edgecx(const _Float16* __restrict__ emid, const float* __restrict__ e1o,
                               const float* __restrict__ e2o, _Float16* __restrict__ out, int b0) {
  int t = blockIdx.x * blockDim.x + threadIdx.x;
  if (t >= ROWS_E * 40) return;
  const int r = t / 40, c4 = t % 40;
  v4h h = {};
  if (c4 < 32) {
    h = *(const v4h*)(emid + (size_t)r * D_ + c4 * 4);
  } else if (c4 < 34) {
    const float* src;
    if (r < ROWS1) { const int b = r / N_, i = r % N_; src = e1o + ((size_t)(b0 + b) * N_ + i) * 9; }
    else { const int rr = r - ROWS1, b = rr / N2_, j = rr % N2_; src = e2o + ((size_t)(b0 + b) * N2_ + j) * 9; }
    const int base = 1 + (c4 - 32) * 4;
    h = cvt4(src[base], src[base + 1], src[base + 2], src[base + 3]);
  }
  *(v4h*)(out + (size_t)r * 160 + c4 * 4) = h;
}

__global__ void k_build_kin1(const _Float16* __restrict__ nodeF, const _Float16* __restrict__ edgeF,
                             const float* __restrict__ gf, const float* __restrict__ tw,
                             const float* __restrict__ tb, _Float16* __restrict__ out, int b0) {
  int t = blockIdx.x * blockDim.x + threadIdx.x;
  if (t >= ROWS2 * 96) return;
  const int r = t / 96, c4 = t % 96;
  v4h h;
  if (c4 < 32) {
    h = *(const v4h*)(nodeF + (size_t)(CB * 21 + r) * D_ + c4 * 4);
  } else if (c4 < 64) {
    h = *(const v4h*)(edgeF + (size_t)(ROWS1 + r) * D_ + (c4 - 32) * 4);
  } else {
    const int b = r / N2_, i = (r % N2_) / N_, j = r % N_;
    const float t1 = fmaxf(0.f, gf[(size_t)(b0 + b) * GF_ + 841 + i]);
    const float t2 = fmaxf(0.f, gf[(size_t)(b0 + b) * GF_ + 861 + i * N_ + j]);
    const float dt = t1 - t2;
    const int d = (c4 - 64) * 4;
    h = cvt4(cosf(dt * tw[d] + tb[d]), cosf(dt * tw[d + 1] + tb[d + 1]),
             cosf(dt * tw[d + 2] + tb[d + 2]), cosf(dt * tw[d + 3] + tb[d + 3]));
  }
  *(v4h*)(out + (size_t)r * 384 + c4 * 4) = h;
}

__global__ void k_build_qin(const _Float16* __restrict__ srcF, const float* __restrict__ tb,
                            _Float16* __restrict__ out, int rows) {
  int t = blockIdx.x * blockDim.x + threadIdx.x;
  if (t >= rows * 64) return;
  const int r = t / 64, c4 = t % 64;
  v4h h;
  if (c4 < 32) {
    h = *(const v4h*)(srcF + (size_t)r * D_ + c4 * 4);
  } else {
    const int d = (c4 - 32) * 4;
    h = cvt4(cosf(tb[d]), cosf(tb[d + 1]), cosf(tb[d + 2]), cosf(tb[d + 3]));
  }
  *(v4h*)(out + (size_t)r * 256 + c4 * 4) = h;
}

__global__ void k_build_min(const _Float16* __restrict__ o, const _Float16* __restrict__ srcF,
                            _Float16* __restrict__ out, int rows) {
  int t = blockIdx.x * blockDim.x + threadIdx.x;
  if (t >= rows * 96) return;
  const int r = t / 96, c4 = t % 96;
  v4h h;
  if (c4 < 64) h = *(const v4h*)(o + (size_t)r * 256 + c4 * 4);
  else         h = *(const v4h*)(srcF + (size_t)r * D_ + (c4 - 64) * 4);
  *(v4h*)(out + (size_t)r * 384 + c4 * 4) = h;
}

__global__ void k_build_kin2(const _Float16* __restrict__ h1p, const _Float16* __restrict__ edgeF,
                             const float* __restrict__ gf, const float* __restrict__ smaxp,
                             const float* __restrict__ tw, const float* __restrict__ tb,
                             _Float16* __restrict__ out, int b0) {
  int t = blockIdx.x * blockDim.x + threadIdx.x;
  if (t >= ROWS1 * 96) return;
  const int r = t / 96, c4 = t % 96;
  v4h h;
  if (c4 < 32) {
    h = *(const v4h*)(h1p + (size_t)(b0 * N_ + r) * D_ + c4 * 4);
  } else if (c4 < 64) {
    h = *(const v4h*)(edgeF + (size_t)r * D_ + (c4 - 32) * 4);
  } else {
    const int b = r / N_, i = r % N_;
    const float ts = fmaxf(0.f, smaxp[b0 + b]);
    const float t1 = fmaxf(0.f, gf[(size_t)(b0 + b) * GF_ + 841 + i]);
    const float dt = ts - t1;
    const int d = (c4 - 64) * 4;
    h = cvt4(cosf(dt * tw[d] + tb[d]), cosf(dt * tw[d + 1] + tb[d + 1]),
             cosf(dt * tw[d + 2] + tb[d + 2]), cosf(dt * tw[d + 3] + tb[d + 3]));
  }
  *(v4h*)(out + (size_t)r * 384 + c4 * 4) = h;
}

// ---------------------------------------------------------------------------
// Softmax attention combine: one wave per (row, head); 20 neighbors.
// ---------------------------------------------------------------------------
__global__ __launch_bounds__(32)
void k_attn_combine(const _Float16* __restrict__ q, const _Float16* __restrict__ k,
                    const _Float16* __restrict__ v, _Float16* __restrict__ o, int rows) {
  const int wid = blockIdx.x;
  if (wid >= rows * 2) return;
  const int r = wid >> 1, h = wid & 1;
  const int lane = threadIdx.x;
  v4h qv = *(const v4h*)(q + (size_t)r * 256 + h * D_ + lane * 4);
  const float q0 = (float)qv[0], q1 = (float)qv[1], q2 = (float)qv[2], q3 = (float)qv[3];
  float s[N_];
#pragma unroll
  for (int j = 0; j < N_; ++j) {
    v4h kv = *(const v4h*)(k + ((size_t)r * N_ + j) * 256 + h * D_ + lane * 4);
    float p = q0 * (float)kv[0] + q1 * (float)kv[1] + q2 * (float)kv[2] + q3 * (float)kv[3];
    for (int off = 16; off; off >>= 1) p += __shfl_xor(p, off, 32);
    s[j] = p * INV_SQRT_DH;
  }
  float mx = s[0];
#pragma unroll
  for (int j = 1; j < N_; ++j) mx = fmaxf(mx, s[j]);
  float sum = 0.f;
#pragma unroll
  for (int j = 0; j < N_; ++j) { s[j] = expf(s[j] - mx); sum += s[j]; }
  const float inv = 1.0f / sum;
  float o0 = 0.f, o1 = 0.f, o2 = 0.f, o3 = 0.f;
#pragma unroll
  for (int j = 0; j < N_; ++j) {
    const float a = s[j] * inv;
    v4h vv = *(const v4h*)(v + ((size_t)r * N_ + j) * 256 + h * D_ + lane * 4);
    o0 += a * (float)vv[0]; o1 += a * (float)vv[1]; o2 += a * (float)vv[2]; o3 += a * (float)vv[3];
  }
  *(v4h*)(o + (size_t)r * 256 + h * D_ + lane * 4) = cvt4(o0, o1, o2, o3);
}

// Build L = concat(hs[:,None,:], h1)  [B,21,D] (f16)
__global__ void k_build_L(const _Float16* __restrict__ hs, const _Float16* __restrict__ h1p,
                          _Float16* __restrict__ L) {
  int t = blockIdx.x * blockDim.x + threadIdx.x;
  if (t >= B_ * 21 * 32) return;
  const int c4 = t % 32, rk = t / 32;
  const int b = rk / 21, kidx = rk % 21;
  v4h h;
  if (kidx == 0) h = *(const v4h*)(hs + (size_t)b * D_ + c4 * 4);
  else           h = *(const v4h*)(h1p + ((size_t)b * N_ + kidx - 1) * D_ + c4 * 4);
  *(v4h*)(L + ((size_t)b * 21 + kidx) * D_ + c4 * 4) = h;
}

// Co-attention: one wave per batch row, 21 keys; f32 output to d_out.
__global__ __launch_bounds__(32)
void k_coatt(const _Float16* __restrict__ sq, const _Float16* __restrict__ Lk,
             const _Float16* __restrict__ L, float* __restrict__ outv) {
  const int b = blockIdx.x;
  const int lane = threadIdx.x;
  v4h qv = *(const v4h*)(sq + (size_t)b * D_ + lane * 4);
  const float q0 = (float)qv[0], q1 = (float)qv[1], q2 = (float)qv[2], q3 = (float)qv[3];
  float s[21];
#pragma unroll
  for (int j = 0; j < 21; ++j) {
    v4h kv = *(const v4h*)(Lk + ((size_t)b * 21 + j) * D_ + lane * 4);
    float p = q0 * (float)kv[0] + q1 * (float)kv[1] + q2 * (float)kv[2] + q3 * (float)kv[3];
    for (int off = 16; off; off >>= 1) p += __shfl_xor(p, off, 32);
    s[j] = p * INV_SQRT_DH;
  }
  float mx = s[0];
#pragma unroll
  for (int j = 1; j < 21; ++j) mx = fmaxf(mx, s[j]);
  float sum = 0.f;
#pragma unroll
  for (int j = 0; j < 21; ++j) { s[j] = expf(s[j] - mx); sum += s[j]; }
  const float inv = 1.0f / sum;
  float o0 = 0.f, o1 = 0.f, o2 = 0.f, o3 = 0.f;
#pragma unroll
  for (int j = 0; j < 21; ++j) {
    const float a = s[j] * inv;
    v4h lv = *(const v4h*)(L + ((size_t)b * 21 + j) * D_ + lane * 4);
    o0 += a * (float)lv[0]; o1 += a * (float)lv[1]; o2 += a * (float)lv[2]; o3 += a * (float)lv[3];
  }
  float4* op = (float4*)(outv + (size_t)b * D_ + lane * 4);
  *op = make_float4(o0, o1, o2, o3);
}

// ---------------------------------------------------------------------------
extern "C" void kernel_launch(void* const* d_in, const int* in_sizes, int n_in,
                              void* d_out, int out_size, void* d_ws, size_t ws_size,
                              hipStream_t stream) {
  (void)in_sizes; (void)n_in; (void)out_size;
  char* base = (char*)d_ws;
  float* out = (float*)d_out;

  // ---- byte arena ----
  size_t off = 0;
  auto ab = [&](size_t bytes) { off = (off + 255) & ~(size_t)255; size_t o = off; off += bytes; return o; };
  auto hsz = [](size_t n) { return n * sizeof(_Float16); };
  const size_t H1P   = ab(hsz((size_t)2 * B_ * N_ * D_));
  const size_t HSH   = ab(hsz((size_t)2 * B_ * D_));
  const size_t LBUF  = ab(hsz((size_t)2 * B_ * 21 * D_));
  const size_t LKB   = ab(hsz((size_t)2 * B_ * 21 * D_));
  const size_t SQB   = ab(hsz((size_t)2 * B_ * D_));
  const size_t SMAXP = ab((size_t)2 * B_ * sizeof(float));
  const size_t NODEX = ab(hsz((size_t)ROWS_N * 192));
  const size_t NODEH = ab(hsz((size_t)ROWS_N * D_));
  const size_t NODEF = ab(hsz((size_t)ROWS_N * D_));
  const size_t EDGEX = ab(hsz((size_t)ROWS_E * 32));
  const size_t EDGEH = ab(hsz((size_t)ROWS_E * D_));
  const size_t EMID  = ab(hsz((size_t)ROWS_E * D_));
  const size_t ECX   = ab(hsz((size_t)ROWS_E * 160));
  const size_t ECH   = ab(hsz((size_t)ROWS_E * D_));
  const size_t EDGEF = ab(hsz((size_t)ROWS_E * D_));
  const size_t KIN1  = ab(hsz((size_t)ROWS2 * 384));
  const size_t K1    = ab(hsz((size_t)ROWS2 * 256));
  const size_t V1    = ab(hsz((size_t)ROWS2 * 256));
  const size_t QIN1  = ab(hsz((size_t)ROWS1 * 256));
  const size_t Q1    = ab(hsz((size_t)ROWS1 * 256));
  const size_t O1    = ab(hsz((size_t)ROWS1 * 256));
  const size_t OW1   = ab(hsz((size_t)ROWS1 * 256));
  const size_t M1IN  = ab(hsz((size_t)ROWS1 * 384));
  const size_t M1H   = ab(hsz((size_t)ROWS1 * D_));
  const size_t KIN2  = ab(hsz((size_t)ROWS1 * 384));
  const size_t K2    = ab(hsz((size_t)ROWS1 * 256));
  const size_t V2    = ab(hsz((size_t)ROWS1 * 256));
  const size_t QIN2  = ab(hsz((size_t)CB * 256));
  const size_t Q2    = ab(hsz((size_t)CB * 256));
  const size_t O2    = ab(hsz((size_t)CB * 256));
  const size_t OW2   = ab(hsz((size_t)CB * 256));
  const size_t M2IN  = ab(hsz((size_t)CB * 384));
  const size_t M2H   = ab(hsz((size_t)CB * D_));
  // weight fragments
  auto frsz = [](int Kpad, int Ncol) { return (size_t)(Kpad / 32) * (Ncol / 16) * 512; };
  size_t foff = 0;
  auto falloc = [&](size_t nh) { size_t o = foff; foff += (nh + 15) & ~(size_t)15; return o; };
  const size_t F_N1 = falloc(frsz(192, 128)), F_N2 = falloc(frsz(128, 128));
  const size_t F_E1 = falloc(frsz(32, 128)),  F_E2 = falloc(frsz(128, 128));
  const size_t F_C1 = falloc(frsz(160, 128)), F_C2 = falloc(frsz(128, 128));
  size_t F_WQ[2], F_WK[2], F_WV[2], F_WO[2], F_WM1[2], F_WM2[2];
  for (int l = 0; l < 2; ++l) {
    F_WQ[l]  = falloc(frsz(256, 256)); F_WK[l]  = falloc(frsz(384, 256));
    F_WV[l]  = falloc(frsz(384, 256)); F_WO[l]  = falloc(frsz(256, 256));
    F_WM1[l] = falloc(frsz(384, 128)); F_WM2[l] = falloc(frsz(128, 128));
  }
  const size_t F_CQ = falloc(frsz(128, 128)), F_CK = falloc(frsz(128, 128));
  const size_t FRAGBASE = ab(hsz(foff));
  if (ws_size < off) return;  // insufficient scratch

  auto Hp = [&](size_t o) { return (_Float16*)(base + o); };
  _Float16* FR = Hp(FRAGBASE);
  float* smaxAll = (float*)(base + SMAXP);
  auto inF = [&](int i) { return (const float*)d_in[i]; };

  // ---- pack all weights into WMMA B-fragment order ----
  auto pack = [&](const float* w, size_t fo, int K, int Kpad, int Ncol) {
    int total = (Kpad / 32) * (Ncol / 16) * 32;
    k_pack_b<<<(total + 255) / 256, 256, 0, stream>>>(w, FR + fo, K, Kpad, Ncol);
  };
  pack(inF(10), F_N1, 172, 192, 128); pack(inF(12), F_N2, 128, 128, 128);
  pack(inF(14), F_E1, 16, 32, 128);   pack(inF(16), F_E2, 128, 128, 128);
  pack(inF(18), F_C1, 136, 160, 128); pack(inF(20), F_C2, 128, 128, 128);
  for (int l = 0; l < 2; ++l) {
    int bi = 24 + l * 8;
    pack(inF(bi + 0), F_WQ[l], 256, 256, 256);
    pack(inF(bi + 1), F_WK[l], 384, 384, 256);
    pack(inF(bi + 2), F_WV[l], 384, 384, 256);
    pack(inF(bi + 3), F_WO[l], 256, 256, 256);
    pack(inF(bi + 4), F_WM1[l], 384, 384, 128);
    pack(inF(bi + 6), F_WM2[l], 128, 128, 128);
  }
  pack(inF(40), F_CQ, 128, 128, 128); pack(inF(41), F_CK, 128, 128, 128);

  auto gemm = [&](const _Float16* X, size_t fo, const float* bias, _Float16* Yh,
                  float* Yf, int M, int Kpad, int Ncol, int relu) {
    dim3 g((M + 15) / 16, Ncol / 64);
    k_gemm<<<g, 32, 0, stream>>>(X, FR + fo, bias, Yh, Yf, M, Kpad, Ncol, relu);
  };
  const float* tw = inF(22); const float* tb = inF(23);

  for (int side = 0; side < 2; ++side) {
    const int*   ids = (const int*)d_in[side];
    const float* gf  = inF(2 + side);
    const float* e1o = inF(side == 0 ? 4 : 6);
    const float* e2o = inF(side == 0 ? 5 : 7);
    float*     smaxp = smaxAll + side * B_;
    _Float16*  h1p   = Hp(H1P) + (size_t)side * B_ * N_ * D_;
    _Float16*  hsh   = Hp(HSH) + (size_t)side * B_ * D_;
    float*     s0out = out + (size_t)side * B_ * D_;

    k_smax<<<(B_ + 63) / 64, 64, 0, stream>>>(gf, smaxp, out + (size_t)4 * B_ * D_ + side * B_);

    for (int c = 0; c < NCHUNK; ++c) {
      const int b0 = c * CB;
      // node MLP: srcf + f1 + f2
      k_gather_node<<<(ROWS_N * 48 + 255) / 256, 256, 0, stream>>>(inF(8), ids, gf, Hp(NODEX), b0);
      gemm(Hp(NODEX), F_N1, inF(11), Hp(NODEH), nullptr, ROWS_N, 192, 128, 1);
      gemm(Hp(NODEH), F_N2, inF(13), Hp(NODEF), nullptr, ROWS_N, 128, 128, 0);
      // edge MLPs
      k_gather_edge<<<(ROWS_E * 8 + 255) / 256, 256, 0, stream>>>(inF(9), e1o, e2o, Hp(EDGEX), b0);
      gemm(Hp(EDGEX), F_E1, inF(15), Hp(EDGEH), nullptr, ROWS_E, 32, 128, 1);
      gemm(Hp(EDGEH), F_E2, inF(17), Hp(EMID), nullptr, ROWS_E, 128, 128, 0);
      k_build_edgecx<<<(ROWS_E * 40 + 255) / 256, 256, 0, stream>>>(Hp(EMID), e1o, e2o, Hp(ECX), b0);
      gemm(Hp(ECX), F_C1, inF(19), Hp(ECH), nullptr, ROWS_E, 160, 128, 1);
      gemm(Hp(ECH), F_C2, inF(21), Hp(EDGEF), nullptr, ROWS_E, 128, 128, 0);
      // attention layer 1 (over 2-hop)
      k_build_kin1<<<(ROWS2 * 96 + 255) / 256, 256, 0, stream>>>(Hp(NODEF), Hp(EDGEF), gf, tw, tb, Hp(KIN1), b0);
      gemm(Hp(KIN1), F_WK[0], nullptr, Hp(K1), nullptr, ROWS2, 384, 256, 0);
      gemm(Hp(KIN1), F_WV[0], nullptr, Hp(V1), nullptr, ROWS2, 384, 256, 0);
      k_build_qin<<<(ROWS1 * 64 + 255) / 256, 256, 0, stream>>>(Hp(NODEF) + (size_t)CB * D_, tb, Hp(QIN1), ROWS1);
      gemm(Hp(QIN1), F_WQ[0], nullptr, Hp(Q1), nullptr, ROWS1, 256, 256, 0);
      k_attn_combine<<<ROWS1 * 2, 32, 0, stream>>>(Hp(Q1), Hp(K1), Hp(V1), Hp(O1), ROWS1);
      gemm(Hp(O1), F_WO[0], nullptr, Hp(OW1), nullptr, ROWS1, 256, 256, 0);
      k_build_min<<<(ROWS1 * 96 + 255) / 256, 256, 0, stream>>>(Hp(OW1), Hp(NODEF) + (size_t)CB * D_, Hp(M1IN), ROWS1);
      gemm(Hp(M1IN), F_WM1[0], inF(29), Hp(M1H), nullptr, ROWS1, 384, 128, 1);
      gemm(Hp(M1H), F_WM2[0], inF(31), h1p + (size_t)b0 * N_ * D_, nullptr, ROWS1, 128, 128, 0);
      // attention layer 2 (over 1-hop)
      k_build_kin2<<<(ROWS1 * 96 + 255) / 256, 256, 0, stream>>>(h1p, Hp(EDGEF), gf, smaxp, tw, tb, Hp(KIN2), b0);
      gemm(Hp(KIN2), F_WK[1], nullptr, Hp(K2), nullptr, ROWS1, 384, 256, 0);
      gemm(Hp(KIN2), F_WV[1], nullptr, Hp(V2), nullptr, ROWS1, 384, 256, 0);
      k_build_qin<<<(CB * 64 + 255) / 256, 256, 0, stream>>>(Hp(NODEF), tb, Hp(QIN2), CB);
      gemm(Hp(QIN2), F_WQ[1], nullptr, Hp(Q2), nullptr, CB, 256, 256, 0);
      k_attn_combine<<<CB * 2, 32, 0, stream>>>(Hp(Q2), Hp(K2), Hp(V2), Hp(O2), CB);
      gemm(Hp(O2), F_WO[1], nullptr, Hp(OW2), nullptr, CB, 256, 256, 0);
      k_build_min<<<(CB * 96 + 255) / 256, 256, 0, stream>>>(Hp(OW2), Hp(NODEF), Hp(M2IN), CB);
      gemm(Hp(M2IN), F_WM1[1], inF(37), Hp(M2H), nullptr, CB, 384, 128, 1);
      gemm(Hp(M2H), F_WM2[1], inF(39), hsh + (size_t)b0 * D_, s0out + (size_t)b0 * D_, CB, 128, 128, 0);
    }
  }

  // ---- co-attention ----
  for (int side = 0; side < 2; ++side) {
    _Float16* L = Hp(LBUF) + (size_t)side * B_ * 21 * D_;
    k_build_L<<<(B_ * 21 * 32 + 255) / 256, 256, 0, stream>>>(
        Hp(HSH) + (size_t)side * B_ * D_, Hp(H1P) + (size_t)side * B_ * N_ * D_, L);
    gemm(L, F_CK, nullptr, Hp(LKB) + (size_t)side * B_ * 21 * D_, nullptr, B_ * 21, 128, 128, 0);
    gemm(Hp(HSH) + (size_t)side * B_ * D_, F_CQ, nullptr, Hp(SQB) + (size_t)side * B_ * D_,
         nullptr, B_, 128, 128, 0);
  }
  // cs = coatt(s0, dstL); cd = coatt(d0, srcL)
  k_coatt<<<B_, 32, 0, stream>>>(Hp(SQB), Hp(LKB) + (size_t)B_ * 21 * D_,
                                 Hp(LBUF) + (size_t)B_ * 21 * D_, out + (size_t)2 * B_ * D_);
  k_coatt<<<B_, 32, 0, stream>>>(Hp(SQB) + (size_t)B_ * D_, Hp(LKB), Hp(LBUF),
                                 out + (size_t)3 * B_ * D_);
}
